// QuaternionLSTM_79645873537026
// MI455X (gfx1250) — compile-verified, matrix operand order checked by reading, not run on hardware
//
#include <hip/hip_runtime.h>
#include <hip/hip_bf16.h>

// ---------------------------------------------------------------------------
// Quaternion LSTM (2 layers) for MI455X / gfx1250 (wave32, WMMA).
//
// qlinear == real GEMM against an expanded weight matrix implementing the
// Hamilton-product structure. All GEMMs run on v_wmma_f32_16x16x32_f16
// (f16 in, f32 accumulate), software-pipelined so global_load_b128 clauses
// overlap the WMMA pipe. The sequential recurrence is 512 small WMMA GEMMs
// + pointwise gate kernels per layer.
// ---------------------------------------------------------------------------

typedef __attribute__((ext_vector_type(16))) _Float16 v16h;
typedef __attribute__((ext_vector_type(8)))  float    v8f;

#define BQ   32      // batch
#define SQ   512     // seq len
#define HQ   256     // hidden (quaternion units)
#define KQ   1024    // flattened input dim = 256*4
#define NQ   4096    // flattened gate-output dim = 4 gates * 256 * 4
#define MPROJ (BQ*SQ) // 16384 rows for the input projection GEMM

// ---- WMMA fragment helpers -------------------------------------------------
// 16-bit A-matrix 16x32 layout (ISA 7.12.2): lane L holds row m=L&15,
// K = k0..k0+7 and k0+16..k0+23 where k0 = (L>>4)*8. Two 16B loads per frag.
__device__ __forceinline__ v16h load_frag(const _Float16* __restrict__ p) {
    union { uint4 u[2]; v16h h; } c;
    c.u[0] = *reinterpret_cast<const uint4*>(p);
    c.u[1] = *reinterpret_cast<const uint4*>(p + 16);
    return c.h;
}

__device__ __forceinline__ v8f wmma_f16(v16h a, v16h b, v8f c) {
    return __builtin_amdgcn_wmma_f32_16x16x32_f16(false, a, false, b,
                                                  (short)0, c, false, false);
}

// ---- small utility kernels -------------------------------------------------

__global__ void k_f32_to_f16(const float* __restrict__ in,
                             _Float16* __restrict__ out, int n) {
    int i = blockIdx.x * blockDim.x + threadIdx.x;
    if (i < n) out[i] = (_Float16)in[i];
}

// Expand quaternion weight W(4,H,In,4) -> real f16 matrix Wreal[N=4096][K=1024]
// row n = g*1024 + o*4 + cout ; col k = i*4 + cin
__global__ void k_expand_w(const float* __restrict__ W,
                           _Float16* __restrict__ Wr) {
    int idx = blockIdx.x * blockDim.x + threadIdx.x;   // 4096*1024 threads
    if (idx >= NQ * KQ) return;
    int n = idx >> 10, k = idx & 1023;
    int g = n >> 10, rem = n & 1023, o = rem >> 2, cout = rem & 3;
    int i = k >> 2, cin = k & 3;
    // Hamilton left-multiplication matrix M(w): y = M(w) x
    const int   comp[4][4] = {{0,1,2,3},{1,0,3,2},{2,3,0,1},{3,2,1,0}};
    const float sgn [4][4] = {{1.f,-1.f,-1.f,-1.f},
                              {1.f, 1.f,-1.f, 1.f},
                              {1.f, 1.f, 1.f,-1.f},
                              {1.f,-1.f, 1.f, 1.f}};
    float w = W[(((size_t)g * HQ + o) * HQ + i) * 4 + comp[cout][cin]];
    Wr[(size_t)n * KQ + k] = (_Float16)(w * sgn[cout][cin]);
}

// bias_comb[n] = bx[g,o,c] + bh[g,o,c]   (both constant over time steps)
__global__ void k_bias(const float* __restrict__ bx,
                       const float* __restrict__ bh,
                       float* __restrict__ bc) {
    int n = blockIdx.x * blockDim.x + threadIdx.x;
    if (n >= NQ) return;
    int g = n >> 10, rem = n & 1023;               // rem = o*4 + c
    bc[n] = bx[g * 1024 + rem] + bh[g * 1024 + rem];
}

__global__ void k_zero_state(_Float16* __restrict__ h16,
                             float* __restrict__ cst) {
    int i = blockIdx.x * blockDim.x + threadIdx.x;  // 32768
    if (i < BQ * KQ) { h16[i] = (_Float16)0.f; cst[i] = 0.f; }
}

// ---- input-projection GEMM: C[M,4096] = A[M,1024] * Wr^T + bias ------------
// Block: 256 thr (8 waves). Block tile 64(M) x 256(N); wave tile 32 x 64
// (2 A-frags x 4 B-frags -> 8 WMMA per k-step). Software-pipelined: next
// k-step's fragments are in flight while the current WMMAs execute.
// Grid: (M/64, N/256) = (256, 16).
__global__ void k_wmma_proj(const _Float16* __restrict__ A,
                            const _Float16* __restrict__ Wr,
                            const float* __restrict__ bias,
                            float* __restrict__ C) {
    int wave = threadIdx.x >> 5, lane = threadIdx.x & 31;
    int m0 = blockIdx.x * 64 + (wave & 1) * 32;
    int n0 = blockIdx.y * 256 + (wave >> 1) * 64;
    int row  = lane & 15;
    int rsel = (lane >> 4) << 3;

    const _Float16* Ap0 = A  + (size_t)(m0 + row) * KQ + rsel;
    const _Float16* Ap1 = Ap0 + (size_t)16 * KQ;
    const _Float16* Wp  = Wr + (size_t)(n0 + row) * KQ + rsel;

    v8f acc[2][4];
#pragma unroll
    for (int i = 0; i < 2; ++i)
#pragma unroll
        for (int j = 0; j < 4; ++j) acc[i][j] = (v8f){};

    // prologue: fragments for kb = 0
    v16h a0 = load_frag(Ap0), a1 = load_frag(Ap1);
    v16h b0 = load_frag(Wp);
    v16h b1 = load_frag(Wp + (size_t)16 * KQ);
    v16h b2 = load_frag(Wp + (size_t)32 * KQ);
    v16h b3 = load_frag(Wp + (size_t)48 * KQ);

#pragma unroll 2
    for (int kb = 0; kb < KQ - 32; kb += 32) {
        // issue next k-step's loads before consuming current fragments
        int kn = kb + 32;
        __builtin_prefetch(Ap0 + kb + 256, 0, 0);   // A stream -> L2
        __builtin_prefetch(Ap1 + kb + 256, 0, 0);
        v16h na0 = load_frag(Ap0 + kn), na1 = load_frag(Ap1 + kn);
        v16h nb0 = load_frag(Wp + kn);
        v16h nb1 = load_frag(Wp + kn + (size_t)16 * KQ);
        v16h nb2 = load_frag(Wp + kn + (size_t)32 * KQ);
        v16h nb3 = load_frag(Wp + kn + (size_t)48 * KQ);

        acc[0][0] = wmma_f16(a0, b0, acc[0][0]);
        acc[0][1] = wmma_f16(a0, b1, acc[0][1]);
        acc[0][2] = wmma_f16(a0, b2, acc[0][2]);
        acc[0][3] = wmma_f16(a0, b3, acc[0][3]);
        acc[1][0] = wmma_f16(a1, b0, acc[1][0]);
        acc[1][1] = wmma_f16(a1, b1, acc[1][1]);
        acc[1][2] = wmma_f16(a1, b2, acc[1][2]);
        acc[1][3] = wmma_f16(a1, b3, acc[1][3]);

        a0 = na0; a1 = na1; b0 = nb0; b1 = nb1; b2 = nb2; b3 = nb3;
    }
    // epilogue k-step
    acc[0][0] = wmma_f16(a0, b0, acc[0][0]);
    acc[0][1] = wmma_f16(a0, b1, acc[0][1]);
    acc[0][2] = wmma_f16(a0, b2, acc[0][2]);
    acc[0][3] = wmma_f16(a0, b3, acc[0][3]);
    acc[1][0] = wmma_f16(a1, b0, acc[1][0]);
    acc[1][1] = wmma_f16(a1, b1, acc[1][1]);
    acc[1][2] = wmma_f16(a1, b2, acc[1][2]);
    acc[1][3] = wmma_f16(a1, b3, acc[1][3]);

    // C/D layout: element v -> m = v + 8*(lane>>4), n = lane&15
    int mb = (lane >> 4) * 8, nn = lane & 15;
#pragma unroll
    for (int i = 0; i < 2; ++i) {
#pragma unroll
        for (int v = 0; v < 8; ++v) {
            size_t r = (size_t)(m0 + 16 * i + mb + v) * NQ;
#pragma unroll
            for (int j = 0; j < 4; ++j)
                C[r + n0 + 16 * j + nn] = acc[i][j][v] + bias[n0 + 16 * j + nn];
        }
    }
}

// ---- recurrent step GEMM: pre[32,4096] = xp_t + h[32,1024] * Wh^T ----------
// 512 independent 16x16 tiles spread over 64 blocks (8 waves each),
// software-pipelined over K. Wh (8 MB f16) and h (64 KB) stay L2-resident.
__global__ void k_wmma_step(const _Float16* __restrict__ h16,
                            const _Float16* __restrict__ Whr,
                            const float* __restrict__ xp, int t,
                            float* __restrict__ pre) {
    int wave = threadIdx.x >> 5, lane = threadIdx.x & 31;
    int tile = blockIdx.x * 8 + wave;      // 0..511
    int m0 = (tile >> 8) * 16;             // 0 or 16
    int n0 = (tile & 255) * 16;
    int row = lane & 15, rsel = (lane >> 4) << 3;

    const _Float16* Ap = h16 + (size_t)(m0 + row) * KQ + rsel;
    const _Float16* Wp = Whr + (size_t)(n0 + row) * KQ + rsel;

    v8f acc = {};
    v16h af = load_frag(Ap), bf = load_frag(Wp);
#pragma unroll 4
    for (int kb = 0; kb < KQ - 32; kb += 32) {
        v16h naf = load_frag(Ap + kb + 32);
        v16h nbf = load_frag(Wp + kb + 32);
        acc = wmma_f16(af, bf, acc);
        af = naf; bf = nbf;
    }
    acc = wmma_f16(af, bf, acc);

    int mb = (lane >> 4) * 8, n = n0 + (lane & 15);
#pragma unroll
    for (int v = 0; v < 8; ++v) {
        int b = m0 + mb + v;                       // b == batch row
        float val = acc[v] + xp[((size_t)b * SQ + t) * NQ + n];
        pre[(size_t)b * NQ + n] = val;
    }
}

// ---- pointwise: gates, Hamilton products, LayerNorm(4) ---------------------
struct f4 { float x, y, z, w; };
__device__ __forceinline__ f4 ld4(const float* p) {
    float4 v = *reinterpret_cast<const float4*>(p);
    return {v.x, v.y, v.z, v.w};
}
__device__ __forceinline__ void st4(float* p, f4 a) {
    *reinterpret_cast<float4*>(p) = make_float4(a.x, a.y, a.z, a.w);
}
__device__ __forceinline__ f4 sig4(f4 a) {
    return { 1.f/(1.f+__expf(-a.x)), 1.f/(1.f+__expf(-a.y)),
             1.f/(1.f+__expf(-a.z)), 1.f/(1.f+__expf(-a.w)) };
}
__device__ __forceinline__ f4 tanh4(f4 a) {
    return { tanhf(a.x), tanhf(a.y), tanhf(a.z), tanhf(a.w) };
}
__device__ __forceinline__ f4 hprod(f4 p, f4 q) {      // Hamilton product
    return { p.x*q.x - p.y*q.y - p.z*q.z - p.w*q.w,
             p.x*q.y + p.y*q.x + p.z*q.w - p.w*q.z,
             p.x*q.z - p.y*q.w + p.z*q.x + p.w*q.y,
             p.x*q.w + p.y*q.z - p.z*q.y + p.w*q.x };
}
__device__ __forceinline__ f4 lnorm4(f4 a, const float* w, const float* b) {
    float mu = 0.25f * (a.x + a.y + a.z + a.w);
    float dx = a.x-mu, dy = a.y-mu, dz = a.z-mu, dw = a.w-mu;
    float var = 0.25f * (dx*dx + dy*dy + dz*dz + dw*dw);
    float r = rsqrtf(var + 1e-5f);
    return { dx*r*w[0]+b[0], dy*r*w[1]+b[1], dz*r*w[2]+b[2], dw*r*w[3]+b[3] };
}

__global__ void k_lstm_point(const float* __restrict__ pre,
                             const float* __restrict__ ln,   // (4,4)
                             float* __restrict__ cst,        // (B,H,4)
                             _Float16* __restrict__ h16,     // (B,1024)
                             _Float16* __restrict__ out16,   // seq f16 or null
                             float* __restrict__ out32,      // seq f32 or null
                             float* __restrict__ hn,         // (B,H,4)
                             float* __restrict__ cn,
                             int t, int writeFinal) {
    int idx = blockIdx.x * blockDim.x + threadIdx.x;   // 0..8191
    if (idx >= BQ * HQ) return;
    int b = idx >> 8, o = idx & 255;
    const float* p = pre + (size_t)b * NQ + o * 4;
    f4 gi = sig4 (ld4(p + 0 * 1024));
    f4 gf = sig4 (ld4(p + 1 * 1024));
    f4 gg = tanh4(ld4(p + 2 * 1024));
    f4 go = sig4 (ld4(p + 3 * 1024));

    f4 cprev = ld4(cst + (size_t)idx * 4);
    f4 fc = hprod(gf, cprev), ig = hprod(gi, gg);
    f4 csum = { fc.x+ig.x, fc.y+ig.y, fc.z+ig.z, fc.w+ig.w };
    f4 cnew = lnorm4(csum, ln + 0, ln + 4);
    f4 hh   = lnorm4(hprod(go, tanh4(cnew)), ln + 8, ln + 12);

    st4(cst + (size_t)idx * 4, cnew);
    // packed 4x f16 (8B) store for next step's GEMM input
    union { _Float16 h[4]; uint2 u; } pk;
    pk.h[0]=(_Float16)hh.x; pk.h[1]=(_Float16)hh.y;
    pk.h[2]=(_Float16)hh.z; pk.h[3]=(_Float16)hh.w;
    *reinterpret_cast<uint2*>(h16 + (size_t)idx * 4) = pk.u;

    size_t srow = ((size_t)b * SQ + t) * KQ + o * 4;
    if (out16) *reinterpret_cast<uint2*>(out16 + srow) = pk.u;
    if (out32) st4(out32 + srow, hh);
    if (writeFinal) {
        st4(hn + (size_t)idx * 4, hh);
        st4(cn + (size_t)idx * 4, cnew);
    }
}

// ---------------------------------------------------------------------------
// Host launcher
// ---------------------------------------------------------------------------
extern "C" void kernel_launch(void* const* d_in, const int* in_sizes, int n_in,
                              void* d_out, int out_size, void* d_ws, size_t ws_size,
                              hipStream_t stream) {
    (void)in_sizes; (void)n_in; (void)out_size; (void)ws_size;
    const float* x   = (const float*)d_in[0];
    const float* Wx0 = (const float*)d_in[1];
    const float* bx0 = (const float*)d_in[2];
    const float* Wh0 = (const float*)d_in[3];
    const float* bh0 = (const float*)d_in[4];
    const float* ln0 = (const float*)d_in[5];
    const float* Wx1 = (const float*)d_in[6];
    const float* bx1 = (const float*)d_in[7];
    const float* Wh1 = (const float*)d_in[8];
    const float* bh1 = (const float*)d_in[9];
    const float* ln1 = (const float*)d_in[10];
    float* out = (float*)d_out;

    char* w = (char*)d_ws;
    float*    xp    = (float*)   (w + 0);                       // 256 MB
    _Float16* xf16  = (_Float16*)(w + 268435456ull);            //  32 MB
    _Float16* hs0   = (_Float16*)(w + 301989888ull);            //  32 MB
    _Float16* WxR   = (_Float16*)(w + 335544320ull);            //   8 MB
    _Float16* WhR   = (_Float16*)(w + 343932928ull);            //   8 MB
    float*    bias  = (float*)   (w + 352321536ull);
    float*    pre   = (float*)   (w + 352337920ull);
    _Float16* h16   = (_Float16*)(w + 352862208ull);
    float*    cst   = (float*)   (w + 352927744ull);

    float* hn0 = out + (size_t)MPROJ * KQ;          // 16777216
    float* hn1 = hn0 + BQ * HQ * 4;
    float* cn0 = hn1 + BQ * HQ * 4;
    float* cn1 = cn0 + BQ * HQ * 4;

    dim3 projGrid(MPROJ / 64, NQ / 256);

    // ================= layer 0 =================
    k_f32_to_f16<<<(MPROJ * KQ) / 256, 256, 0, stream>>>(x, xf16, MPROJ * KQ);
    k_expand_w  <<<(NQ * KQ) / 256, 256, 0, stream>>>(Wx0, WxR);
    k_expand_w  <<<(NQ * KQ) / 256, 256, 0, stream>>>(Wh0, WhR);
    k_bias      <<<NQ / 256, 256, 0, stream>>>(bx0, bh0, bias);
    k_wmma_proj <<<projGrid, 256, 0, stream>>>(xf16, WxR, bias, xp);
    k_zero_state<<<(BQ * KQ) / 256, 256, 0, stream>>>(h16, cst);
    for (int t = 0; t < SQ; ++t) {
        k_wmma_step <<<64, 256, 0, stream>>>(h16, WhR, xp, t, pre);
        k_lstm_point<<<BQ * HQ / 256, 256, 0, stream>>>(
            pre, ln0, cst, h16, hs0, nullptr, hn0, cn0, t, t == SQ - 1);
    }

    // ================= layer 1 =================
    k_expand_w  <<<(NQ * KQ) / 256, 256, 0, stream>>>(Wx1, WxR);
    k_expand_w  <<<(NQ * KQ) / 256, 256, 0, stream>>>(Wh1, WhR);
    k_bias      <<<NQ / 256, 256, 0, stream>>>(bx1, bh1, bias);
    k_wmma_proj <<<projGrid, 256, 0, stream>>>(hs0, WxR, bias, xp);
    k_zero_state<<<(BQ * KQ) / 256, 256, 0, stream>>>(h16, cst);
    for (int t = 0; t < SQ; ++t) {
        k_wmma_step <<<64, 256, 0, stream>>>(h16, WhR, xp, t, pre);
        k_lstm_point<<<BQ * HQ / 256, 256, 0, stream>>>(
            pre, ln1, cst, h16, nullptr, out, hn1, cn1, t, t == SQ - 1);
    }
}
